// HSLSTMRegressor_45981919871025
// MI455X (gfx1250) — compile-verified, matrix operand order checked by reading, not run on hardware
//
#include <hip/hip_runtime.h>
#include <hip/hip_bf16.h>

// ---------------------------------------------------------------------------
// Types for WMMA fragments (gfx1250, wave32)
// ---------------------------------------------------------------------------
typedef __bf16 v8bf  __attribute__((ext_vector_type(8)));
typedef __bf16 v16bf __attribute__((ext_vector_type(16)));
typedef float  v8f   __attribute__((ext_vector_type(8)));

// Scheduler shaping (CK-style): force "loads of next tile before WMMAs of
// current tile" so the allocator cannot collapse the software pipeline.
#if __has_builtin(__builtin_amdgcn_sched_group_barrier)
#define SCHED_GROUP(mask, size) __builtin_amdgcn_sched_group_barrier((mask), (size), 0)
#else
#define SCHED_GROUP(mask, size)
#endif
#define SG_VMEM_READ 0x020  // VMEM read instructions
#define SG_WMMA      0x008  // MFMA/WMMA instructions

static __device__ __forceinline__ v16bf frag_cat(v8bf lo, v8bf hi) {
    return __builtin_shufflevector(lo, hi, 0,1,2,3,4,5,6,7,8,9,10,11,12,13,14,15);
}

// Load a 16-bit WMMA fragment half-pair for this lane.
// Per CDNA5 ISA 16-bit A/B layout: lane's VGPR0..3 hold K = kbase+0..7,
// VGPR4..7 hold K = kbase+16..23 (kbase already includes +8 for lanes 16..31).
static __device__ __forceinline__ v16bf load_frag(const __bf16* __restrict__ base) {
    const v8bf* p = (const v8bf*)base;
    v8bf lo = p[0];   // K kbase+0  .. kbase+7
    v8bf hi = p[2];   // K kbase+16 .. kbase+23
    return frag_cat(lo, hi);
}

static __device__ __forceinline__ float sigmoidf_fast(float x) {
    return 1.0f / (1.0f + __expf(-x));
}

static __device__ __forceinline__ v8f wmma_bf16(v16bf a, v16bf b, v8f c) {
    return __builtin_amdgcn_wmma_f32_16x16x32_bf16(
               false, a, false, b, (short)0, c, false, false);
}

// ---------------------------------------------------------------------------
// One K-segment of the gate GEMM, software-pipelined with prefetch depth 1,
// pinned by sched_group_barriers:
//   [V10] ; { [V10][W4] } x (NT-1) ; [W4]
// => loads(t0), loads(t1), wmma(t0), loads(t2), wmma(t1), ... : every WMMA
// group waits only behind the NEXT tile's in-flight loads.
// ---------------------------------------------------------------------------
template<int NT, int GOFF>
static __device__ __forceinline__ void k_segment(
    const __bf16* __restrict__ pa,   // lane's A base (K contiguous, +32/tile)
    const __bf16* __restrict__ pb,   // lane's B base (K contiguous, +32/tile)
    v8f (&acc)[4])
{
    v16bf a  = load_frag(pa);
    v16bf b0 = load_frag(pb);
    v16bf b1 = load_frag(pb + GOFF);
    v16bf b2 = load_frag(pb + 2 * GOFF);
    v16bf b3 = load_frag(pb + 3 * GOFF);
    SCHED_GROUP(SG_VMEM_READ, 10);          // tile 0 loads

#pragma unroll
    for (int i = 0; i < NT - 1; ++i) {
        const __bf16* qa = pa + (i + 1) * 32;
        const __bf16* qb = pb + (i + 1) * 32;
        const v16bf a_n = load_frag(qa);
        const v16bf c0  = load_frag(qb);
        const v16bf c1  = load_frag(qb + GOFF);
        const v16bf c2  = load_frag(qb + 2 * GOFF);
        const v16bf c3  = load_frag(qb + 3 * GOFF);

        acc[0] = wmma_bf16(a, b0, acc[0]);
        acc[1] = wmma_bf16(a, b1, acc[1]);
        acc[2] = wmma_bf16(a, b2, acc[2]);
        acc[3] = wmma_bf16(a, b3, acc[3]);

        SCHED_GROUP(SG_VMEM_READ, 10);      // tile i+1 loads
        SCHED_GROUP(SG_WMMA, 4);            // tile i WMMAs

        a = a_n; b0 = c0; b1 = c1; b2 = c2; b3 = c3;
    }
    // drain: last tile has no prefetch
    acc[0] = wmma_bf16(a, b0, acc[0]);
    acc[1] = wmma_bf16(a, b1, acc[1]);
    acc[2] = wmma_bf16(a, b2, acc[2]);
    acc[3] = wmma_bf16(a, b3, acc[3]);
    SCHED_GROUP(SG_WMMA, 4);
}

// ---------------------------------------------------------------------------
// Fused LSTM layer step:
//   gates = [A0 | A1] @ Bw^T + bias  (bf16 WMMA, f32 accum), LSTM cell update
//   fused in the epilogue. Each wave owns a 16(M) x 16(n) tile accumulated
//   for ALL FOUR gates (weight rows n + g*512).
// Grid: (H/16 = 32, B/64 = 4), 128 threads (4 waves stacked along M).
// ---------------------------------------------------------------------------
template<int LDA0, int K0, int K1>
__global__ __launch_bounds__(128) void lstm_layer_step(
    const __bf16* __restrict__ A0,   // [256, LDA0], K [0, K0)
    const __bf16* __restrict__ A1,   // [256, 512],  K [K0, K0+K1)
    const __bf16* __restrict__ Bw,   // [2048, KTOT] packed bf16 row-major
    const float*  __restrict__ bias, // [2048]
    float*  __restrict__ c,          // [256,512] in/out
    __bf16* __restrict__ h_out,      // [256,512] bf16 (next-step WMMA input)
    float*  __restrict__ hf_out)     // optional f32 copy (final step, layer1)
{
    constexpr int H    = 512;
    constexpr int KTOT = K0 + K1;
    constexpr int GOFF = H * KTOT;   // element offset between gates in Bw

    const int lane  = threadIdx.x & 31;
    const int wave  = threadIdx.x >> 5;
    const int l15   = lane & 15;
    const int khalf = lane >> 4;              // 0: K 0..7/16..23, 1: +8
    const int m0 = blockIdx.y * 64 + wave * 16;
    const int n  = blockIdx.x * 16 + l15;     // this lane's hidden column

    // Accumulators: gate g in {i,f,g,o} lives at weight row n + g*H.
    v8f acc[4];
#pragma unroll
    for (int g = 0; g < 4; ++g) {
        const float bv = bias[n + g * H];
#pragma unroll
        for (int v = 0; v < 8; ++v) acc[g][v] = bv;
    }

    const __bf16* pb = Bw + (size_t)n * KTOT + khalf * 8;

    // Segment 1: A0 covers K [0, K0)
    k_segment<K0 / 32, GOFF>(
        A0 + (size_t)(m0 + l15) * LDA0 + khalf * 8, pb, acc);

    // Segment 2: A1 covers K [K0, KTOT)
    k_segment<K1 / 32, GOFF>(
        A1 + (size_t)(m0 + l15) * H + khalf * 8, pb + K0, acc);

    // Fused LSTM cell epilogue. C/D layout: lane holds one N column,
    // VGPR v -> M = m0 + 8*khalf + v. PyTorch gate order: i, f, g, o.
#pragma unroll
    for (int v = 0; v < 8; ++v) {
        const int m = m0 + khalf * 8 + v;
        const size_t idx = (size_t)m * H + n;
        const float ig = sigmoidf_fast(acc[0][v]);
        const float fg = sigmoidf_fast(acc[1][v]);
        const float gg = tanhf(acc[2][v]);
        const float og = sigmoidf_fast(acc[3][v]);
        const float cn = fg * c[idx] + ig * gg;
        const float hn = og * tanhf(cn);
        c[idx]     = cn;
        h_out[idx] = (__bf16)hn;
        if (hf_out) hf_out[idx] = hn;
    }
}

// ---------------------------------------------------------------------------
// Final head: out[b] = h1[b,:] . W_lin + b_lin ; loss = mean((out-y)^2)
// ---------------------------------------------------------------------------
__global__ __launch_bounds__(256) void final_head(
    const float* __restrict__ h1,
    const float* __restrict__ Wl,
    const float* __restrict__ bl,
    const float* __restrict__ y,
    float* __restrict__ out)   // 256 outputs + 1 loss
{
    __shared__ float red[256];
    const int b = threadIdx.x;
    float acc = bl[0];
    for (int k = 0; k < 512; ++k) acc += h1[(size_t)b * 512 + k] * Wl[k];
    out[b] = acc;
    const float d = acc - y[b];
    red[b] = d * d;
    __syncthreads();
    for (int s = 128; s > 0; s >>= 1) {
        if (b < s) red[b] += red[b + s];
        __syncthreads();
    }
    if (b == 0) out[256] = red[0] * (1.0f / 256.0f);
}

// ---------------------------------------------------------------------------
// One-time prep kernels
// ---------------------------------------------------------------------------
__global__ __launch_bounds__(256) void cvt_bf16(const float* __restrict__ in,
                                                __bf16* __restrict__ out, long n)
{
    const long i = (long)blockIdx.x * 256 + threadIdx.x;
    if (i < n) out[i] = (__bf16)in[i];
}

// Pack W = [Wa | Wb] (concat along K) to bf16 row-major [rows, Ka+Kb];
// also fuse bias = ba + bb.
__global__ __launch_bounds__(256) void pack_w(
    const float* __restrict__ Wa, int Ka,
    const float* __restrict__ Wb, int Kb,
    const float* __restrict__ ba, const float* __restrict__ bb,
    __bf16* __restrict__ Bw, float* __restrict__ bias, int rows)
{
    const int  Kt  = Ka + Kb;
    const long idx = (long)blockIdx.x * 256 + threadIdx.x;
    const long total = (long)rows * Kt;
    if (idx < total) {
        const int nn = (int)(idx / Kt);
        const int k  = (int)(idx % Kt);
        const float v = (k < Ka) ? Wa[(long)nn * Ka + k]
                                 : Wb[(long)nn * Kb + (k - Ka)];
        Bw[idx] = (__bf16)v;
    }
    if (idx < rows) bias[idx] = ba[idx] + bb[idx];
}

__global__ __launch_bounds__(256) void zero_state(
    float* __restrict__ c0, float* __restrict__ c1,
    __bf16* __restrict__ h0, __bf16* __restrict__ h1,   // double buffers (2x BH)
    float* __restrict__ h1f, int BH)
{
    const int i = blockIdx.x * 256 + threadIdx.x;   // < BH
    c0[i] = 0.0f; c1[i] = 0.0f; h1f[i] = 0.0f;
    h0[i] = (__bf16)0.0f; h0[i + BH] = (__bf16)0.0f;
    h1[i] = (__bf16)0.0f; h1[i + BH] = (__bf16)0.0f;
}

// ---------------------------------------------------------------------------
// Host-side launch
// ---------------------------------------------------------------------------
extern "C" void kernel_launch(void* const* d_in, const int* in_sizes, int n_in,
                              void* d_out, int out_size, void* d_ws, size_t ws_size,
                              hipStream_t stream)
{
    constexpr int B = 256, T = 512, I = 64, H = 512;
    constexpr int G = 4 * H;            // 2048
    constexpr int K0tot = I + H;        // 576
    constexpr int K1tot = H + H;        // 1024
    constexpr int BH = B * H;

    const float* x     = (const float*)d_in[0];
    const float* y     = (const float*)d_in[1];
    const float* W_ih0 = (const float*)d_in[2];
    const float* W_hh0 = (const float*)d_in[3];
    const float* b_ih0 = (const float*)d_in[4];
    const float* b_hh0 = (const float*)d_in[5];
    const float* W_ih1 = (const float*)d_in[6];
    const float* W_hh1 = (const float*)d_in[7];
    const float* b_ih1 = (const float*)d_in[8];
    const float* b_hh1 = (const float*)d_in[9];
    const float* W_lin = (const float*)d_in[10];
    const float* b_lin = (const float*)d_in[11];
    float* out = (float*)d_out;

    // Workspace carve-out (256B aligned)
    char* ws = (char*)d_ws;
    size_t off = 0;
    auto carve = [&](size_t bytes) -> void* {
        void* p = ws + off;
        off = (off + bytes + 255) & ~(size_t)255;
        return p;
    };
    __bf16* xbf   = (__bf16*)carve((size_t)B * T * I * 2);   // 16 MB
    __bf16* Bw0   = (__bf16*)carve((size_t)G * K0tot * 2);   // 2.25 MB
    __bf16* Bw1   = (__bf16*)carve((size_t)G * K1tot * 2);   // 4 MB
    float*  bias0 = (float*)carve((size_t)G * 4);
    float*  bias1 = (float*)carve((size_t)G * 4);
    __bf16* h0bf  = (__bf16*)carve((size_t)2 * BH * 2);      // double buffer
    __bf16* h1bf  = (__bf16*)carve((size_t)2 * BH * 2);      // double buffer
    float*  c0    = (float*)carve((size_t)BH * 4);
    float*  c1    = (float*)carve((size_t)BH * 4);
    float*  h1f   = (float*)carve((size_t)BH * 4);
    (void)ws_size; (void)in_sizes; (void)n_in; (void)out_size;

    // ---- one-time prep ----
    zero_state<<<BH / 256, 256, 0, stream>>>(c0, c1, h0bf, h1bf, h1f, BH);

    const long xN = (long)B * T * I;
    cvt_bf16<<<(unsigned)((xN + 255) / 256), 256, 0, stream>>>(x, xbf, xN);

    const long w0N = (long)G * K0tot;
    pack_w<<<(unsigned)((w0N + 255) / 256), 256, 0, stream>>>(
        W_ih0, I, W_hh0, H, b_ih0, b_hh0, Bw0, bias0, G);

    const long w1N = (long)G * K1tot;
    pack_w<<<(unsigned)((w1N + 255) / 256), 256, 0, stream>>>(
        W_ih1, H, W_hh1, H, b_ih1, b_hh1, Bw1, bias1, G);

    // ---- recurrent loop: 2 fused kernels per step, double-buffered h ----
    const dim3 step_grid(H / 16, B / 64);   // (32, 4)
    for (int t = 0; t < T; ++t) {
        const int r = t & 1;        // read buffer
        const int w = r ^ 1;        // write buffer

        // Layer 0: K = [x_t (64) | h0_prev (512)]
        lstm_layer_step<T * I, I, H><<<step_grid, 128, 0, stream>>>(
            xbf + (size_t)t * I,
            h0bf + (size_t)r * BH,
            Bw0, bias0,
            c0, h0bf + (size_t)w * BH, nullptr);

        // Layer 1: K = [h0_new (512) | h1_prev (512)]
        lstm_layer_step<H, H, H><<<step_grid, 128, 0, stream>>>(
            h0bf + (size_t)w * BH,
            h1bf + (size_t)r * BH,
            Bw1, bias1,
            c1, h1bf + (size_t)w * BH, (t == T - 1) ? h1f : nullptr);
    }

    // ---- output head: out[256] + loss ----
    final_head<<<1, 256, 0, stream>>>(h1f, W_lin, b_lin, y, out);
}
